// DmTranslatePredict_54820962566566
// MI455X (gfx1250) — matrix operand order repaired and checked by reading, not compiled, not run on hardware
//
#include <hip/hip_runtime.h>
#include <cmath>

// ---------------------------------------------------------------------------
// GNMT-style translate predict for MI455X (gfx1250, wave32, WMMA).
// Strategy:
//   * convert all weights to f16 once, repacked into WMMA B-fragment layout
//   * encoder input projection X@Wx as one big parallel WMMA GEMM (+bias)
//   * 128 sequential fused LSTM-recurrent steps (h@Wh WMMA + gates)
//   * keys = memory @ mem_W as one WMMA GEMM
//   * 48 decode steps: fused WMMA LSTM (K=1280 concat), dot-attention
//     (VALU, negligible FLOPs), attn projection WMMA GEMM, fused
//     logits WMMA GEMM + per-tile shfl argmax, final argmax reduce.
// ---------------------------------------------------------------------------

#define B_   32
#define S_   128
#define U_   512
#define E_   256
#define TGTV 32000
#define NSTEPS 48

typedef __attribute__((ext_vector_type(16))) _Float16 v16h;
typedef __attribute__((ext_vector_type(8)))  float    v8f;

union FragU { v16h v; float4 q[2]; };

// A fragment (16x32 f16, row-major source, stride lda in elements).
// Lane l holds row m0+(l&15); K runs [16*hs, 16*hs+16).
__device__ __forceinline__ v16h load_a_frag(const _Float16* __restrict__ A,
                                            int lda, int m0, int k0, int lane) {
  int row = m0 + (lane & 15);
  int hs  = lane >> 4;
  const _Float16* base = A + (size_t)row * lda + k0 + 16 * hs;
  FragU u;
  u.q[0] = *(const float4*)(base);
  u.q[1] = *(const float4*)(base + 8);
  return u.v;
}

// B fragment from pre-packed weights: tile = 32 lanes x 16 f16, lane-major.
__device__ __forceinline__ v16h load_b_frag(const _Float16* __restrict__ Bp,
                                            size_t tile, int lane) {
  const _Float16* base = Bp + (tile * 32 + (size_t)lane) * 16;
  FragU u;
  u.q[0] = ((const float4*)base)[0];
  u.q[1] = ((const float4*)base)[1];
  return u.v;
}

__device__ __forceinline__ v8f wmma32(v16h a, v16h b, v8f c) {
  return __builtin_amdgcn_wmma_f32_16x16x32_f16(false, a, false, b,
                                                (short)0, c, false, false);
}

__device__ __forceinline__ float sigmf_(float x) { return 1.0f / (1.0f + __expf(-x)); }

// ---------------------------------------------------------------------------
// Weight repack: fp32 row-major [K,N] -> f16 WMMA-B tiles.
// Packed element idx: e=idx&15, lane=(idx>>4)&31, tile=idx>>9,
// kT=tile%kTiles, nT=tile/kTiles;  K=kT*32+(lane>>4)*16+e, N=nT*16+(lane&15).
// ---------------------------------------------------------------------------
__global__ void pack_w_kernel(const float* __restrict__ W, _Float16* __restrict__ out,
                              int K, int N) {
  int idx = blockIdx.x * blockDim.x + threadIdx.x;
  if (idx >= K * N) return;
  int e    = idx & 15;
  int lane = (idx >> 4) & 31;
  int tile = idx >> 9;
  int kTiles = K >> 5;
  int kT = tile % kTiles;
  int nT = tile / kTiles;
  int kk = kT * 32 + (lane >> 4) * 16 + e;
  int nn = nT * 16 + (lane & 15);
  out[idx] = (_Float16)W[(size_t)kk * N + nn];
}

// dec_Wx (768x2048) stacked over dec_Wh (512x2048) -> packed [1280,2048]
__global__ void pack_decw_kernel(const float* __restrict__ Wx,
                                 const float* __restrict__ Wh,
                                 _Float16* __restrict__ out) {
  const int K = 1280, N = 2048, kTiles = 40;
  int idx = blockIdx.x * blockDim.x + threadIdx.x;
  if (idx >= K * N) return;
  int e    = idx & 15;
  int lane = (idx >> 4) & 31;
  int tile = idx >> 9;
  int kT = tile % kTiles;
  int nT = tile / kTiles;
  int kk = kT * 32 + (lane >> 4) * 16 + e;
  int nn = nT * 16 + (lane & 15);
  float v = (kk < 768) ? Wx[(size_t)kk * N + nn]
                       : Wh[(size_t)(kk - 768) * N + nn];
  out[idx] = (_Float16)v;
}

// Encoder embedding gather -> f16 X [B*S, 256]
__global__ void embed_enc_kernel(const int* __restrict__ tokens,
                                 const float* __restrict__ table,
                                 _Float16* __restrict__ X) {
  int idx = blockIdx.x * blockDim.x + threadIdx.x; // B*S*E
  if (idx >= B_ * S_ * E_) return;
  int row = idx >> 8;
  int e   = idx & 255;
  X[idx] = (_Float16)table[(size_t)tokens[row] * E_ + e];
}

__global__ void zero_hc_kernel(_Float16* __restrict__ h0, float* __restrict__ c) {
  int idx = blockIdx.x * blockDim.x + threadIdx.x;
  if (idx < B_ * U_) { h0[idx] = (_Float16)0.0f; c[idx] = 0.0f; }
}

// ---------------------------------------------------------------------------
// Generic WMMA GEMM: C[M,N](f32) = A[M,K] @ Bpacked (+ bias[N]).
// One wave per 16x16 C tile.
// ---------------------------------------------------------------------------
__global__ void gemm_f32out_kernel(const _Float16* __restrict__ A, int lda,
                                   const _Float16* __restrict__ Bp,
                                   const float* __restrict__ bias,
                                   float* __restrict__ C, int ldc,
                                   int mTiles, int nTiles, int kTiles) {
  int gid  = blockIdx.x * (blockDim.x >> 5) + (threadIdx.x >> 5);
  int lane = threadIdx.x & 31;
  if (gid >= mTiles * nTiles) return;
  int mT = gid % mTiles, nT = gid / mTiles;
  int m0 = mT * 16, n0 = nT * 16;
  int col = n0 + (lane & 15);
  int hs  = lane >> 4;
  float b0 = bias ? bias[col] : 0.0f;
  v8f acc;
  for (int r = 0; r < 8; ++r) acc[r] = b0;
  for (int kt = 0; kt < kTiles; ++kt) {
    v16h a = load_a_frag(A, lda, m0, kt * 32, lane);
    v16h b = load_b_frag(Bp, (size_t)nT * kTiles + kt, lane);
    acc = wmma32(a, b, acc);
  }
  for (int r = 0; r < 8; ++r)
    C[(size_t)(m0 + r + 8 * hs) * ldc + col] = acc[r];
}

// Same GEMM but f16 output with stride (used for attn -> next cell_in slot).
__global__ void gemm_f16out_kernel(const _Float16* __restrict__ A, int lda,
                                   const _Float16* __restrict__ Bp,
                                   _Float16* __restrict__ C, int ldc,
                                   int mTiles, int nTiles, int kTiles) {
  int gid  = blockIdx.x * (blockDim.x >> 5) + (threadIdx.x >> 5);
  int lane = threadIdx.x & 31;
  if (gid >= mTiles * nTiles) return;
  int mT = gid % mTiles, nT = gid / mTiles;
  int m0 = mT * 16, n0 = nT * 16;
  int col = n0 + (lane & 15);
  int hs  = lane >> 4;
  v8f acc;
  for (int r = 0; r < 8; ++r) acc[r] = 0.0f;
  for (int kt = 0; kt < kTiles; ++kt) {
    v16h a = load_a_frag(A, lda, m0, kt * 32, lane);
    v16h b = load_b_frag(Bp, (size_t)nT * kTiles + kt, lane);
    acc = wmma32(a, b, acc);
  }
  for (int r = 0; r < 8; ++r)
    C[(size_t)(m0 + r + 8 * hs) * ldc + col] = (_Float16)acc[r];
}

// ---------------------------------------------------------------------------
// Encoder recurrent step: z = XW[:,t,:] (bias folded) + h@Wh ; gates ; h,c.
// 64 waves: wave -> (mT in 0..1, uT in 0..31); each wave does 4 gate tiles.
// h is double-buffered (hin/hout) to avoid cross-wave RAW races.
// ---------------------------------------------------------------------------
__global__ void enc_lstm_step_kernel(const _Float16* __restrict__ hin,
                                     _Float16* __restrict__ hout,
                                     const _Float16* __restrict__ WhP,
                                     const float* __restrict__ XW,
                                     float* __restrict__ c,
                                     float* __restrict__ hf32,
                                     float* __restrict__ memory,
                                     _Float16* __restrict__ memf16,
                                     int t) {
  int gid  = blockIdx.x * (blockDim.x >> 5) + (threadIdx.x >> 5);
  int lane = threadIdx.x & 31;
  int mT = gid & 1, uT = gid >> 1;
  int m0 = mT * 16, n0 = uT * 16;
  int hs = lane >> 4;
  int col = n0 + (lane & 15);
  const int KT = 16; // K = 512
  v8f acc[4];
  for (int g = 0; g < 4; ++g)
    for (int r = 0; r < 8; ++r) {
      int row = m0 + r + 8 * hs;
      acc[g][r] = XW[((size_t)row * S_ + t) * 2048 + g * U_ + col];
    }
  for (int kt = 0; kt < KT; ++kt) {
    v16h a = load_a_frag(hin, U_, m0, kt * 32, lane);
    for (int g = 0; g < 4; ++g) {
      v16h b = load_b_frag(WhP, (size_t)(g * 32 + uT) * KT + kt, lane);
      acc[g] = wmma32(a, b, acc[g]);
    }
  }
  for (int r = 0; r < 8; ++r) {
    int row = m0 + r + 8 * hs;
    size_t idx = (size_t)row * U_ + col;
    float iv = sigmf_(acc[0][r]);
    float fv = sigmf_(acc[1][r]);
    float gv = tanhf(acc[2][r]);
    float ov = sigmf_(acc[3][r]);
    float cn = fv * c[idx] + iv * gv;
    float hn = ov * tanhf(cn);
    c[idx] = cn;
    hf32[idx] = hn;
    hout[idx] = (_Float16)hn;
    size_t midx = ((size_t)row * S_ + t) * U_ + col;
    memory[midx] = hn;
    memf16[midx] = (_Float16)hn;
  }
}

// Decoder fused LSTM: A = decIn_cur [32,1280] = [embed|attn|h], K=1280.
// Writes new h into hc[:,0:512] (attn GEMM A) and decIn_next[:,768:1280].
__global__ void dec_lstm_step_kernel(const _Float16* __restrict__ decInCur,
                                     const _Float16* __restrict__ WP,
                                     const float* __restrict__ bias, // dec_b[2048]
                                     float* __restrict__ c,
                                     float* __restrict__ hf32,
                                     _Float16* __restrict__ hc,
                                     _Float16* __restrict__ decInNext) {
  int gid  = blockIdx.x * (blockDim.x >> 5) + (threadIdx.x >> 5);
  int lane = threadIdx.x & 31;
  int mT = gid & 1, uT = gid >> 1;
  int m0 = mT * 16, n0 = uT * 16;
  int hs = lane >> 4;
  int col = n0 + (lane & 15);
  const int KT = 40; // K = 1280
  v8f acc[4];
  for (int g = 0; g < 4; ++g) {
    float b0 = bias[g * U_ + col];
    for (int r = 0; r < 8; ++r) acc[g][r] = b0;
  }
  for (int kt = 0; kt < KT; ++kt) {
    v16h a = load_a_frag(decInCur, 1280, m0, kt * 32, lane);
    for (int g = 0; g < 4; ++g) {
      v16h b = load_b_frag(WP, (size_t)(g * 32 + uT) * KT + kt, lane);
      acc[g] = wmma32(a, b, acc[g]);
    }
  }
  for (int r = 0; r < 8; ++r) {
    int row = m0 + r + 8 * hs;
    size_t idx = (size_t)row * U_ + col;
    float iv = sigmf_(acc[0][r]);
    float fv = sigmf_(acc[1][r]);
    float gv = tanhf(acc[2][r]);
    float ov = sigmf_(acc[3][r]);
    float cn = fv * c[idx] + iv * gv;
    float hn = ov * tanhf(cn);
    c[idx] = cn;
    hf32[idx] = hn;
    hc[(size_t)row * 1024 + col] = (_Float16)hn;
    decInNext[(size_t)row * 1280 + 768 + col] = (_Float16)hn;
  }
}

// score[b,s] = h[b] . keys[b,s]
__global__ void score_kernel(const float* __restrict__ h,
                             const float* __restrict__ keys,
                             float* __restrict__ score) {
  int idx = blockIdx.x * blockDim.x + threadIdx.x; // 4096
  if (idx >= B_ * S_) return;
  int b = idx >> 7, s = idx & 127;
  const float* hp = h + (size_t)b * U_;
  const float* kp = keys + ((size_t)b * S_ + s) * U_;
  float acc = 0.0f;
  for (int u = 0; u < U_; ++u) acc = fmaf(hp[u], kp[u], acc);
  score[idx] = acc;
}

// in-place softmax over S=128 per batch row; one wave per row.
__global__ void softmax_kernel(float* __restrict__ score) {
  int b    = threadIdx.x >> 5;
  int lane = threadIdx.x & 31;
  float4 v = *(float4*)(score + (size_t)b * S_ + lane * 4);
  float mx = fmaxf(fmaxf(v.x, v.y), fmaxf(v.z, v.w));
  for (int m = 1; m < 32; m <<= 1) mx = fmaxf(mx, __shfl_xor(mx, m, 32));
  float e0 = __expf(v.x - mx), e1 = __expf(v.y - mx);
  float e2 = __expf(v.z - mx), e3 = __expf(v.w - mx);
  float s = e0 + e1 + e2 + e3;
  for (int m = 1; m < 32; m <<= 1) s += __shfl_xor(s, m, 32);
  float inv = 1.0f / s;
  float4 o; o.x = e0 * inv; o.y = e1 * inv; o.z = e2 * inv; o.w = e3 * inv;
  *(float4*)(score + (size_t)b * S_ + lane * 4) = o;
}

// ctx[b,u] = sum_s align[b,s]*memory[b,s,u]  -> hc[:,512:1024] as f16
__global__ void ctx_kernel(const float* __restrict__ align,
                           const float* __restrict__ memory,
                           _Float16* __restrict__ hc) {
  int idx = blockIdx.x * blockDim.x + threadIdx.x; // 16384
  if (idx >= B_ * U_) return;
  int b = idx >> 9, u = idx & 511;
  const float* al = align + (size_t)b * S_;
  const float* mp = memory + (size_t)b * S_ * U_ + u;
  float acc = 0.0f;
  for (int s = 0; s < S_; ++s) acc = fmaf(al[s], mp[(size_t)s * U_], acc);
  hc[(size_t)b * 1024 + 512 + u] = (_Float16)acc;
}

// logits = attn @ out_W + out_b, fused with per-16-col-tile argmax.
// A = decIn_next+256 (lda 1280). 4000 waves = 2 mTiles x 2000 nTiles.
__global__ void logits_argmax_part_kernel(const _Float16* __restrict__ A, int lda,
                                          const _Float16* __restrict__ WoP,
                                          const float* __restrict__ outb,
                                          float* __restrict__ pVal,
                                          int* __restrict__ pIdx) {
  int gid  = blockIdx.x * (blockDim.x >> 5) + (threadIdx.x >> 5);
  int lane = threadIdx.x & 31;
  int mT = gid & 1;
  int nT = gid >> 1; // 0..1999
  int m0 = mT * 16, n0 = nT * 16;
  int hs = lane >> 4;
  int col = n0 + (lane & 15);
  const int KT = 16;
  v8f acc;
  float b0 = outb[col];
  for (int r = 0; r < 8; ++r) acc[r] = b0;
  for (int kt = 0; kt < KT; ++kt) {
    v16h a = load_a_frag(A, lda, m0, kt * 32, lane);
    v16h b = load_b_frag(WoP, (size_t)nT * KT + kt, lane);
    acc = wmma32(a, b, acc);
  }
  // per row r: max over the 16 lanes of this half-wave (cols n0..n0+15)
  for (int r = 0; r < 8; ++r) {
    float val = acc[r];
    int   cand = col;
    for (int m = 1; m <= 8; m <<= 1) {
      float ov = __shfl_xor(val, m, 32);
      int   oc = __shfl_xor(cand, m, 32);
      if (ov > val || (ov == val && oc < cand)) { val = ov; cand = oc; }
    }
    if ((lane & 15) == 0) {
      int row = m0 + r + 8 * hs;
      pVal[(size_t)row * 2000 + nT] = val;
      pIdx[(size_t)row * 2000 + nT] = cand;
    }
  }
}

__global__ void argmax_reduce_kernel(const float* __restrict__ pVal,
                                     const int* __restrict__ pIdx,
                                     int* __restrict__ tok,
                                     float* __restrict__ outSample,
                                     int step) {
  __shared__ float sv[256];
  __shared__ int   si[256];
  int b = blockIdx.x;
  int tid = threadIdx.x;
  float best = -__builtin_inff();
  int bi = 0x7fffffff;
  for (int t = tid; t < 2000; t += 256) {
    float v = pVal[(size_t)b * 2000 + t];
    int   ix = pIdx[(size_t)b * 2000 + t];
    if (v > best || (v == best && ix < bi)) { best = v; bi = ix; }
  }
  sv[tid] = best; si[tid] = bi;
  __syncthreads();
  for (int off = 128; off > 0; off >>= 1) {
    if (tid < off) {
      if (sv[tid + off] > sv[tid] ||
          (sv[tid + off] == sv[tid] && si[tid + off] < si[tid])) {
        sv[tid] = sv[tid + off]; si[tid] = si[tid + off];
      }
    }
    __syncthreads();
  }
  if (tid == 0) {
    tok[b] = si[0];
    outSample[(size_t)b * NSTEPS + step] = (float)si[0];
  }
}

// Decoder init: attn0 = 0, copy encoder-final h (f16) into cell-in slot, tok=START.
__global__ void dec_init_kernel(_Float16* __restrict__ decIn0,
                                const _Float16* __restrict__ hFinal,
                                int* __restrict__ tok) {
  int idx = blockIdx.x * blockDim.x + threadIdx.x; // 32*1024
  if (idx < B_) tok[idx] = 1; // START_TOK
  if (idx >= B_ * 1024) return;
  int row = idx >> 10, cc = idx & 1023;
  if (cc < 512) decIn0[(size_t)row * 1280 + 256 + cc] = (_Float16)0.0f;
  else          decIn0[(size_t)row * 1280 + 768 + (cc - 512)] =
                    hFinal[(size_t)row * U_ + (cc - 512)];
}

// decoder embedding gather -> decIn_next[:, 0:256]
__global__ void embed_dec_kernel(const int* __restrict__ tok,
                                 const float* __restrict__ table,
                                 _Float16* __restrict__ decInNext) {
  int idx = blockIdx.x * blockDim.x + threadIdx.x; // 32*256
  if (idx >= B_ * E_) return;
  int b = idx >> 8, e = idx & 255;
  decInNext[(size_t)b * 1280 + e] = (_Float16)table[(size_t)tok[b] * E_ + e];
}

// ---------------------------------------------------------------------------
extern "C" void kernel_launch(void* const* d_in, const int* in_sizes, int n_in,
                              void* d_out, int out_size, void* d_ws, size_t ws_size,
                              hipStream_t stream) {
  (void)in_sizes; (void)n_in; (void)out_size; (void)ws_size;
  const int*   inputs    = (const int*)d_in[0];
  const float* enc_embed = (const float*)d_in[1];
  const float* enc_Wx    = (const float*)d_in[2];
  const float* enc_Wh    = (const float*)d_in[3];
  const float* enc_b     = (const float*)d_in[4];
  const float* dec_embed = (const float*)d_in[5];
  const float* dec_Wx    = (const float*)d_in[6];
  const float* dec_Wh    = (const float*)d_in[7];
  const float* dec_b     = (const float*)d_in[8];
  const float* mem_W     = (const float*)d_in[9];
  const float* attn_W    = (const float*)d_in[10];
  const float* out_W     = (const float*)d_in[11];
  const float* out_b     = (const float*)d_in[12];
  float* out = (float*)d_out;

  // scratch layout
  char* ws = (char*)d_ws;
  size_t off = 0;
  auto alloc = [&](size_t bytes) -> void* {
    void* p = ws + off;
    off += (bytes + 255) & ~(size_t)255;
    return p;
  };
  _Float16* Xf16    = (_Float16*)alloc((size_t)B_ * S_ * E_ * 2);
  _Float16* encWxP  = (_Float16*)alloc((size_t)E_ * 2048 * 2);
  _Float16* encWhP  = (_Float16*)alloc((size_t)U_ * 2048 * 2);
  _Float16* decWP   = (_Float16*)alloc((size_t)1280 * 2048 * 2);
  _Float16* memWP   = (_Float16*)alloc((size_t)U_ * U_ * 2);
  _Float16* attnWP  = (_Float16*)alloc((size_t)1024 * U_ * 2);
  _Float16* outWP   = (_Float16*)alloc((size_t)U_ * TGTV * 2);
  float*    XW      = (float*)alloc((size_t)B_ * S_ * 2048 * 4);
  float*    memory  = (float*)alloc((size_t)B_ * S_ * U_ * 4);
  _Float16* memf16  = (_Float16*)alloc((size_t)B_ * S_ * U_ * 2);
  float*    keys    = (float*)alloc((size_t)B_ * S_ * U_ * 4);
  float*    hf32    = (float*)alloc((size_t)B_ * U_ * 4);
  float*    cbuf    = (float*)alloc((size_t)B_ * U_ * 4);
  _Float16* hb0     = (_Float16*)alloc((size_t)B_ * U_ * 2);
  _Float16* hb1     = (_Float16*)alloc((size_t)B_ * U_ * 2);
  _Float16* decIn0  = (_Float16*)alloc((size_t)B_ * 1280 * 2);
  _Float16* decIn1  = (_Float16*)alloc((size_t)B_ * 1280 * 2);
  _Float16* hc      = (_Float16*)alloc((size_t)B_ * 1024 * 2);
  float*    score   = (float*)alloc((size_t)B_ * S_ * 4);
  float*    pVal    = (float*)alloc((size_t)B_ * 2000 * 4);
  int*      pIdx    = (int*)alloc((size_t)B_ * 2000 * 4);
  int*      tok     = (int*)alloc((size_t)B_ * 4);
  _Float16* hb[2]     = { hb0, hb1 };
  _Float16* decIn[2]  = { decIn0, decIn1 };

  const int T = 256;
  auto blocks = [](size_t n, int t) { return (int)((n + t - 1) / t); };

  // --- one-time weight conversion / packing -------------------------------
  pack_w_kernel<<<blocks((size_t)E_ * 2048, T), T, 0, stream>>>(enc_Wx, encWxP, E_, 2048);
  pack_w_kernel<<<blocks((size_t)U_ * 2048, T), T, 0, stream>>>(enc_Wh, encWhP, U_, 2048);
  pack_decw_kernel<<<blocks((size_t)1280 * 2048, T), T, 0, stream>>>(dec_Wx, dec_Wh, decWP);
  pack_w_kernel<<<blocks((size_t)U_ * U_, T), T, 0, stream>>>(mem_W, memWP, U_, U_);
  pack_w_kernel<<<blocks((size_t)1024 * U_, T), T, 0, stream>>>(attn_W, attnWP, 1024, U_);
  pack_w_kernel<<<blocks((size_t)U_ * TGTV, T), T, 0, stream>>>(out_W, outWP, U_, TGTV);

  // --- encoder ------------------------------------------------------------
  embed_enc_kernel<<<blocks((size_t)B_ * S_ * E_, T), T, 0, stream>>>(inputs, enc_embed, Xf16);
  zero_hc_kernel<<<blocks((size_t)B_ * U_, T), T, 0, stream>>>(hb[0], cbuf);

  // XW = X @ enc_Wx + enc_b : M=4096, N=2048, K=256 -> 32768 waves
  gemm_f32out_kernel<<<(256 * 128) / 8, T, 0, stream>>>(
      Xf16, E_, encWxP, enc_b, XW, 2048, 256, 128, 8);

  for (int t = 0; t < S_; ++t) {
    enc_lstm_step_kernel<<<8, T, 0, stream>>>(
        hb[t & 1], hb[(t + 1) & 1], encWhP, XW, cbuf, hf32, memory, memf16, t);
  }

  // keys = memory @ mem_W : M=4096, N=512, K=512 -> 8192 waves
  gemm_f32out_kernel<<<(256 * 32) / 8, T, 0, stream>>>(
      memf16, U_, memWP, nullptr, keys, U_, 256, 32, 16);

  // --- decoder ------------------------------------------------------------
  dec_init_kernel<<<blocks((size_t)B_ * 1024, T), T, 0, stream>>>(decIn[0], hb[S_ & 1], tok);
  embed_dec_kernel<<<blocks((size_t)B_ * E_, T), T, 0, stream>>>(tok, dec_embed, decIn[0]);

  for (int step = 0; step < NSTEPS; ++step) {
    _Float16* cur = decIn[step & 1];
    _Float16* nxt = decIn[(step + 1) & 1];

    dec_lstm_step_kernel<<<8, T, 0, stream>>>(cur, decWP, dec_b, cbuf, hf32, hc, nxt);
    score_kernel<<<blocks((size_t)B_ * S_, T), T, 0, stream>>>(hf32, keys, score);
    softmax_kernel<<<1, 1024, 0, stream>>>(score);
    ctx_kernel<<<blocks((size_t)B_ * U_, T), T, 0, stream>>>(score, memory, hc);

    // attn = hc @ attn_W -> f16 into next cell-in slot: M=32,N=512,K=1024
    gemm_f16out_kernel<<<(2 * 32) / 8, T, 0, stream>>>(
        hc, 1024, attnWP, nxt + 256, 1280, 2, 32, 32);

    // logits + per-tile argmax: 2 x 2000 tiles = 4000 waves
    logits_argmax_part_kernel<<<4000 / 8, T, 0, stream>>>(
        nxt + 256, 1280, outWP, out_b, pVal, pIdx);
    argmax_reduce_kernel<<<B_, T, 0, stream>>>(pVal, pIdx, tok, out, step);
    embed_dec_kernel<<<blocks((size_t)B_ * E_, T), T, 0, stream>>>(tok, dec_embed, nxt);
  }
}
